// mSWEGNNLayer_21114059227743
// MI455X (gfx1250) — matrix operand order repaired and checked by reading, not compiled, not run on hardware
//
#include <hip/hip_runtime.h>

// ---------------------------------------------------------------------------
// mSWEGNN layer for MI455X (gfx1250, wave32, WMMA).
//
// Phase 0: zero f32 aggregate buffer (d_ws).
// Phase 1: edge kernel — per-wave 16-edge tile, psi MLP (160->64->64->32) done
//          entirely with v_wmma_f32_16x16x32_bf16 (f32 accumulate), then
//          s_ij = psi * (h_dj - h_di) scattered via global_atomic_add_f32.
// Phase 2: node kernel — out = h_d_prev + agg @ W, again via WMMA.
// ---------------------------------------------------------------------------

typedef __attribute__((ext_vector_type(16))) __bf16   v16bf;
typedef __attribute__((ext_vector_type(8)))  __bf16   v8bf;
typedef __attribute__((ext_vector_type(2)))  __bf16   v2bf;
typedef __attribute__((ext_vector_type(8)))  float    v8f;
typedef __attribute__((ext_vector_type(8)))  unsigned v8u;

#define WAVES_PER_BLOCK 8
#define ACT_STRIDE 72   // 64 + 8 bf16 pad -> 144B row stride, conflict-free b128 reads

// Pack two f32 into one dword of two bf16 (element 0 = lo).
// round-to-nearest via +0x8000, then one v_perm_b32 packs the high halves.
__device__ __forceinline__ unsigned pk_bf16(float lo, float hi) {
  unsigned ulo = __builtin_bit_cast(unsigned, lo) + 0x8000u;
  unsigned uhi = __builtin_bit_cast(unsigned, hi) + 0x8000u;
  return __builtin_amdgcn_perm(uhi, ulo, 0x07060302u);  // D = {uhi[31:16], ulo[31:16]}
}

__device__ __forceinline__ __bf16 f2bf(float f) {
  return __builtin_bit_cast(v2bf, pk_bf16(f, f))[0];
}

// Build a 16-bit A fragment (16x32 tile, this lane = row M, K-halves per ISA
// layout: elements 0..7 -> K = 8*hf + j, elements 8..15 -> K = 16 + 8*hf + j)
// from a row of 32 f32 values in global memory. 4x b128 loads + 12 VALU.
__device__ __forceinline__ v16bf gather_a32(const float* __restrict__ row, int hf) {
  const float4 p0 = *(const float4*)(row + 8 * hf);
  const float4 p1 = *(const float4*)(row + 8 * hf + 4);
  const float4 p2 = *(const float4*)(row + 16 + 8 * hf);
  const float4 p3 = *(const float4*)(row + 16 + 8 * hf + 4);
  v8u u;
  u[0] = pk_bf16(p0.x, p0.y); u[1] = pk_bf16(p0.z, p0.w);
  u[2] = pk_bf16(p1.x, p1.y); u[3] = pk_bf16(p1.z, p1.w);
  u[4] = pk_bf16(p2.x, p2.y); u[5] = pk_bf16(p2.z, p2.w);
  u[6] = pk_bf16(p3.x, p3.y); u[7] = pk_bf16(p3.z, p3.w);
  return __builtin_bit_cast(v16bf, u);
}

__device__ __forceinline__ v8f bias_relu(v8f acc, float b) {
#pragma unroll
  for (int r = 0; r < 8; ++r) {
    float v = acc[r] + b;
    acc[r] = v > 0.f ? v : 0.f;
  }
  return acc;
}

// C/D layout (lane = col N, vgpr r = row M = r + 8*hf) -> LDS row-major bf16.
__device__ __forceinline__ void store_act(__bf16* actp, v8f v, int hf, int l15, int ncol) {
  const int col = ncol * 16 + l15;
#pragma unroll
  for (int r = 0; r < 8; r += 2) {
    v2bf p = __builtin_bit_cast(v2bf, pk_bf16(v[r], v[r + 1]));
    actp[(r + 8 * hf)     * ACT_STRIDE + col] = p[0];
    actp[(r + 1 + 8 * hf) * ACT_STRIDE + col] = p[1];
  }
}

// Read an A fragment (row M = l15, K-step t of 32) back from the LDS staging
// buffer: two aligned 16B chunks per lane.
__device__ __forceinline__ v16bf load_act_a(const __bf16* actp, int l15, int hf, int t) {
  int base = l15 * ACT_STRIDE + t * 32 + 8 * hf;
  v8bf lo = *(const v8bf*)(actp + base);
  v8bf hi = *(const v8bf*)(actp + base + 16);
  v16bf a;
#pragma unroll
  for (int j = 0; j < 8; ++j) { a[j] = lo[j]; a[8 + j] = hi[j]; }
  return a;
}

__global__ void zero_f32(float* __restrict__ p, int n) {
  int i = blockIdx.x * blockDim.x + threadIdx.x;
  int stride = gridDim.x * blockDim.x;
  for (; i < n; i += stride) p[i] = 0.f;
}

// ---------------------------------------------------------------------------
// Edge kernel: 32 WMMA ops per 16-edge tile.
// Fragment table in LDS: frags 0..19  = layer0 (kstep t 0..4 x ntile n 0..3)
//                        frags 20..27 = layer1 (t 0..1 x n 0..3)
//                        frags 28..31 = layer2 (t 0..1 x n 0..1)  [hoisted]
// Each fragment: 32 lanes x 16 bf16 (lane holds N = n*16 + L%16,
// K = t*32 + 8*(L/16) + e, plus 8 for e >= 8) stored contiguously per lane.
// Only layer-2 B is hoisted to registers (32 VGPRs) — hoisting more starves
// the scheduler of registers and serializes the layer-0 gather loads.
// ---------------------------------------------------------------------------
__global__ __launch_bounds__(256) void edge_psi_kernel(
    const float* __restrict__ h_d, const float* __restrict__ h_s,
    const float* __restrict__ ef,  const int* __restrict__ snd,
    const int* __restrict__ rcv,
    const float* __restrict__ w0, const float* __restrict__ b0,
    const float* __restrict__ w1, const float* __restrict__ b1,
    const float* __restrict__ w2, const float* __restrict__ b2,
    float* __restrict__ agg, int E, int nTiles)
{
  __shared__ __align__(16) __bf16 wfrag[32 * 512];
  __shared__ __align__(16) __bf16 act[WAVES_PER_BLOCK * 16 * ACT_STRIDE];

  const int tid = threadIdx.x;

  // ---- stage all psi weights once per block, pre-permuted to B-fragment layout
  for (int i = tid; i < 32 * 512; i += 256) {
    int f = i >> 9, rr = i & 511, L = rr >> 4, e = rr & 15;
    int hfx = L >> 4, l15x = L & 15;
    const float* w; int ncols, t, n;
    if (f < 20)      { t = f >> 2;        n = f & 3;        w = w0; ncols = 64; }
    else if (f < 28) { t = (f - 20) >> 2; n = (f - 20) & 3; w = w1; ncols = 64; }
    else             { t = (f - 28) >> 1; n = (f - 28) & 1; w = w2; ncols = 32; }
    int K = t * 32 + (e < 8 ? 8 * hfx + e : 8 + 8 * hfx + e);
    int N = n * 16 + l15x;
    wfrag[i] = f2bf(w[K * ncols + N]);
  }
  __syncthreads();

  const int lane = tid & 31, l15 = lane & 15, hf = lane >> 4;
  const int wlocal = tid >> 5;
  __bf16* myact = act + wlocal * 16 * ACT_STRIDE;

  const float b0v[4] = { b0[l15], b0[16 + l15], b0[32 + l15], b0[48 + l15] };
  const float b1v[4] = { b1[l15], b1[16 + l15], b1[32 + l15], b1[48 + l15] };
  const float b2v[2] = { b2[l15], b2[16 + l15] };

  // Loop-invariant layer-2 B fragments in registers (32 VGPRs).
  v16bf bw2[4];
#pragma unroll
  for (int f = 0; f < 4; ++f)
    bw2[f] = *(const v16bf*)&wfrag[(28 + f) * 512 + lane * 16];

  const int wave = blockIdx.x * WAVES_PER_BLOCK + wlocal;
  const int nWaves = gridDim.x * WAVES_PER_BLOCK;

  for (int tile = wave; tile < nTiles; tile += nWaves) {
    // prefetch next tile's streamed data (global_prefetch_b8)
    int nextTile = tile + nWaves;
    if (nextTile < nTiles) {
      int pe = nextTile * 16 + l15;
      __builtin_prefetch(&snd[pe], 0, 1);
      __builtin_prefetch(&rcv[pe], 0, 1);
      __builtin_prefetch(ef + (size_t)pe * 32, 0, 1);
    }

    int eM = tile * 16 + l15;
    if (eM >= E) eM = E - 1;                       // tail clamp (no EXEC change)
    const int si = snd[eM], ri = rcv[eM];
    const float* srcs[5] = { h_s + si * 32, h_s + ri * 32,
                             h_d + si * 32, h_d + ri * 32, ef + eM * 32 };

    // ----- layer 0: [16x160] @ [160x64]  (B streamed from LDS)
    v8f acc0[4] = {};
#pragma unroll
    for (int t = 0; t < 5; ++t) {
      v16bf a = gather_a32(srcs[t], hf);
#pragma unroll
      for (int n = 0; n < 4; ++n) {
        v16bf b = *(const v16bf*)&wfrag[(t * 4 + n) * 512 + lane * 16];
        acc0[n] = __builtin_amdgcn_wmma_f32_16x16x32_bf16(
            false, a, false, b, (short)0, acc0[n], false, false);
      }
    }
#pragma unroll
    for (int n = 0; n < 4; ++n) store_act(myact, bias_relu(acc0[n], b0v[n]), hf, l15, n);
    asm volatile("s_wait_dscnt 0" ::: "memory");

    // ----- layer 1: [16x64] @ [64x64]  (B streamed from LDS)
    v8f acc1[4] = {};
#pragma unroll
    for (int t = 0; t < 2; ++t) {
      v16bf a = load_act_a(myact, l15, hf, t);
#pragma unroll
      for (int n = 0; n < 4; ++n) {
        v16bf b = *(const v16bf*)&wfrag[(20 + t * 4 + n) * 512 + lane * 16];
        acc1[n] = __builtin_amdgcn_wmma_f32_16x16x32_bf16(
            false, a, false, b, (short)0, acc1[n], false, false);
      }
    }
    asm volatile("s_wait_dscnt 0" ::: "memory");
#pragma unroll
    for (int n = 0; n < 4; ++n) store_act(myact, bias_relu(acc1[n], b1v[n]), hf, l15, n);
    asm volatile("s_wait_dscnt 0" ::: "memory");

    // ----- layer 2: [16x64] @ [64x32]  (B in registers)
    v8f acc2[2] = {};
#pragma unroll
    for (int t = 0; t < 2; ++t) {
      v16bf a = load_act_a(myact, l15, hf, t);
#pragma unroll
      for (int n = 0; n < 2; ++n)
        acc2[n] = __builtin_amdgcn_wmma_f32_16x16x32_bf16(
            false, a, false, bw2[t * 2 + n], (short)0, acc2[n], false, false);
    }
#pragma unroll
    for (int n = 0; n < 2; ++n) acc2[n] = bias_relu(acc2[n], b2v[n]);

    // ----- s_ij = psi * (h_dj - h_di); scatter-add into aggregate (f32 atomics)
#pragma unroll
    for (int r = 0; r < 8; ++r) {
      int srcL = r + 8 * hf;
      int eIdx = tile * 16 + srcL;
      int sR = __shfl(si, srcL, 32);
      int rR = __shfl(ri, srcL, 32);
      if (eIdx < E) {
#pragma unroll
        for (int n = 0; n < 2; ++n) {
          int feat = n * 16 + l15;
          float diff = h_d[rR * 32 + feat] - h_d[sR * 32 + feat];
          atomicAdd(&agg[rR * 32 + feat], acc2[n][r] * diff);
        }
      }
    }
  }
}

// ---------------------------------------------------------------------------
// Node kernel: out = h_d_prev + agg @ W  (M=16 nodes per wave-tile, K=32, N=32)
// ---------------------------------------------------------------------------
__global__ __launch_bounds__(256) void node_out_kernel(
    const float* __restrict__ h_d, const float* __restrict__ agg,
    const float* __restrict__ W,   float* __restrict__ out,
    int Nn, int nTiles)
{
  const int tid = threadIdx.x;
  const int lane = tid & 31, l15 = lane & 15, hf = lane >> 4;
  const int wave = blockIdx.x * WAVES_PER_BLOCK + (tid >> 5);
  const int nWaves = gridDim.x * WAVES_PER_BLOCK;

  // W is 32x32 -> two B fragments, loop-invariant, kept in registers.
  v16bf bw[2];
#pragma unroll
  for (int n = 0; n < 2; ++n) {
    v8u u;
#pragma unroll
    for (int e = 0; e < 16; e += 2)
      u[e >> 1] = pk_bf16(W[((e < 8 ? 8 * hf + e : 8 + 8 * hf + e)) * 32 + n * 16 + l15],
                          W[((e < 8 ? 8 * hf + e : 8 + 8 * hf + e) + 1) * 32 + n * 16 + l15]);
    bw[n] = __builtin_bit_cast(v16bf, u);
  }

  for (int tile = wave; tile < nTiles; tile += nWaves) {
    int node = tile * 16 + l15;
    if (node >= Nn) node = Nn - 1;
    v16bf a = gather_a32(agg + node * 32, hf);
    v8f acc[2] = {};
#pragma unroll
    for (int n = 0; n < 2; ++n)
      acc[n] = __builtin_amdgcn_wmma_f32_16x16x32_bf16(
          false, a, false, bw[n], (short)0, acc[n], false, false);
#pragma unroll
    for (int r = 0; r < 8; ++r) {
      int nd = tile * 16 + r + 8 * hf;
      if (nd < Nn) {
#pragma unroll
        for (int n = 0; n < 2; ++n) {
          int feat = n * 16 + l15;
          out[nd * 32 + feat] = h_d[nd * 32 + feat] + acc[n][r];
        }
      }
    }
  }
}

extern "C" void kernel_launch(void* const* d_in, const int* in_sizes, int n_in,
                              void* d_out, int out_size, void* d_ws, size_t ws_size,
                              hipStream_t stream) {
  const float* h_d_prev = (const float*)d_in[0];
  const float* h_s      = (const float*)d_in[1];
  const float* ef       = (const float*)d_in[2];
  const int*   snd      = (const int*)d_in[3];
  const int*   rcv      = (const int*)d_in[4];
  const float* w0       = (const float*)d_in[5];
  const float* b0       = (const float*)d_in[6];
  const float* w1       = (const float*)d_in[7];
  const float* b1       = (const float*)d_in[8];
  const float* w2       = (const float*)d_in[9];
  const float* b2       = (const float*)d_in[10];
  const float* W        = (const float*)d_in[11];

  const int E  = in_sizes[3];
  const int Nn = in_sizes[0] / 32;

  float* agg = (float*)d_ws;           // Nn*32 f32 aggregate scratch
  float* out = (float*)d_out;

  // Phase 0: zero the aggregate (d_ws is poisoned, not re-zeroed by harness).
  zero_f32<<<256, 256, 0, stream>>>(agg, Nn * 32);

  // Phase 1: edge MLP + scatter.
  const int nTilesE = (E + 15) / 16;
  int blocksE = (nTilesE + WAVES_PER_BLOCK - 1) / WAVES_PER_BLOCK;
  if (blocksE > 1024) blocksE = 1024;
  edge_psi_kernel<<<blocksE, 256, 0, stream>>>(
      h_d_prev, h_s, ef, snd, rcv, w0, b0, w1, b1, w2, b2, agg, E, nTilesE);

  // Phase 2: out = h_d_prev + agg @ W.
  const int nTilesN = (Nn + 15) / 16;
  int blocksN = (nTilesN + WAVES_PER_BLOCK - 1) / WAVES_PER_BLOCK;
  if (blocksN > 1024) blocksN = 1024;
  node_out_kernel<<<blocksN, 256, 0, stream>>>(h_d_prev, agg, W, out, Nn, nTilesN);
}
